// CodecTransformer_88948772700263
// MI455X (gfx1250) — compile-verified, matrix operand order checked by reading, not run on hardware
//
#include <hip/hip_runtime.h>
#include <hip/hip_bf16.h>

// ---------------- problem constants ----------------
static constexpr int BATCH = 4;
static constexpr int SEQ   = 2048;
static constexpr int DIM   = 1024;   // D == H*HD
static constexpr int NHEAD = 8;
static constexpr int HDIM  = 128;
static constexpr int FFDIM = 4096;
static constexpr int WIN   = 16;     // keys in [i-16, i] -> 17 taps
static constexpr int ROWS  = BATCH * SEQ;   // 8192

typedef __bf16 bf16;
typedef __attribute__((ext_vector_type(16))) __bf16 v16bf;
typedef __attribute__((ext_vector_type(8)))  float  v8f;

__device__ __forceinline__ bf16 f2bf(float x) {
    unsigned u = __builtin_bit_cast(unsigned, x);
    unsigned r = (u + 0x7FFFu + ((u >> 16) & 1u)) >> 16;
    return __builtin_bit_cast(bf16, (unsigned short)r);
}

// ---------------- RMS norm (fp32 in; optional fp32 and/or bf16 out) --------
__global__ __launch_bounds__(256)
void ct_rmsnorm_kernel(const float* __restrict__ in, const float* __restrict__ w,
                       float eps, float* __restrict__ outF, bf16* __restrict__ outH,
                       int D)
{
    const int row = blockIdx.x;
    const float* xr = in + (size_t)row * D;
    __shared__ float red[256];
    float s = 0.f;
    for (int d = threadIdx.x; d < D; d += 256) { float v = xr[d]; s += v * v; }
    red[threadIdx.x] = s;
    __syncthreads();
    for (int st = 128; st > 0; st >>= 1) {
        if (threadIdx.x < st) red[threadIdx.x] += red[threadIdx.x + st];
        __syncthreads();
    }
    const float rstd = rsqrtf(red[0] / (float)D + eps);
    for (int d = threadIdx.x; d < D; d += 256) {
        float v = xr[d] * rstd * w[d];
        if (outF) outF[(size_t)row * D + d] = v;
        if (outH) outH[(size_t)row * D + d] = f2bf(v);
    }
}

// ------------- weight convert fp32 [K,N] -> bf16 transposed [N,K] ----------
__global__ __launch_bounds__(256)
void ct_convt_kernel(const float* __restrict__ W, bf16* __restrict__ Wt, int K, int N)
{
    size_t idx = (size_t)blockIdx.x * 256 + threadIdx.x;
    if (idx >= (size_t)K * N) return;
    int k = (int)(idx / N), n = (int)(idx % N);
    Wt[(size_t)n * K + k] = f2bf(W[idx]);
}

// ---------------- WMMA bf16 GEMM: C[M,N] = A[M,K] * Bt[N,K]^T --------------
// block tile 128x128, 8 waves in 4x2 grid, wave tile 32x64 = 2x4 WMMA frags.
// Double-buffered LDS; tiles staged with global_load_async_to_lds_b128
// (ASYNCcnt) so the stage of tile k+1 overlaps WMMA on tile k.
enum { CT_MODE_F32 = 0, CT_MODE_RESID = 1, CT_MODE_SWIGLU = 2 };

__global__ __launch_bounds__(256)
void ct_gemm_bf16_kernel(const bf16* __restrict__ A, const bf16* __restrict__ Bt,
                         int M, int N, int K, int mode,
                         const float* __restrict__ aux,    // resid (RESID) or g1 (SWIGLU)
                         const float* __restrict__ scale,  // per-N channel scale (RESID)
                         float* __restrict__ outF, bf16* __restrict__ outH)
{
    __shared__ alignas(16) bf16 As[2][128 * 32];
    __shared__ alignas(16) bf16 Bs[2][128 * 32];

    const int tid  = threadIdx.x;
    const int lane = tid & 31, wid = tid >> 5;
    const int wm = wid >> 1, wn = wid & 1;       // wave grid 4 x 2
    const int half = lane >> 4, l16 = lane & 15;
    const int bm = blockIdx.y * 128;
    const int bn = blockIdx.x * 128;

    // per-thread staging coordinates: 128 rows x 32 cols, 16 bf16 per thread
    const int sr  = tid >> 1;            // 0..127
    const int sc16 = (tid & 1) * 16;     // 0 or 16

    v8f c[2][4] = {};

    // async stage of one 128x32 A tile + 128x32 Bt tile into buffer `buf`
    auto stage = [&](int kk, int buf) {
        const unsigned long long ga =
            (unsigned long long)&A[(size_t)(bm + sr) * K + kk + sc16];
        const unsigned long long gb =
            (unsigned long long)&Bt[(size_t)(bn + sr) * K + kk + sc16];
        const unsigned la = (unsigned)(unsigned long long)&As[buf][sr * 32 + sc16];
        const unsigned lb = (unsigned)(unsigned long long)&Bs[buf][sr * 32 + sc16];
        // INST_OFFSET is added to BOTH the LDS and the global address.
        asm volatile(
            "global_load_async_to_lds_b128 %0, %2, off\n\t"
            "global_load_async_to_lds_b128 %0, %2, off offset:16\n\t"
            "global_load_async_to_lds_b128 %1, %3, off\n\t"
            "global_load_async_to_lds_b128 %1, %3, off offset:16"
            :: "v"(la), "v"(lb), "v"(ga), "v"(gb)
            : "memory");
    };

    stage(0, 0);
    asm volatile("s_wait_asynccnt 0x0" ::: "memory");
    __syncthreads();

    for (int k0 = 0; k0 < K; k0 += 32) {
        const int cur = (k0 >> 5) & 1;
        if (k0 + 32 < K) stage(k0 + 32, cur ^ 1);
        if (k0 + 64 < K) {   // pull the K+2 tile toward L2 (global_prefetch_b8)
            __builtin_prefetch(&A[(size_t)(bm + sr) * K + k0 + 64], 0, 1);
            __builtin_prefetch(&Bt[(size_t)(bn + sr) * K + k0 + 64], 0, 1);
        }

        // fragment loads follow the 16-bit A layout:
        // lanes 0-15: K = 0..7,16..23 ; lanes 16-31: K = 8..15,24..31
        v16bf af[2], bfm[4];
#pragma unroll
        for (int mi = 0; mi < 2; ++mi) {
            const bf16* s = &As[cur][(wm * 32 + mi * 16 + l16) * 32 + half * 8];
            *((int4*)&af[mi])     = *(const int4*)(s);
            *((int4*)&af[mi] + 1) = *(const int4*)(s + 16);
        }
#pragma unroll
        for (int ni = 0; ni < 4; ++ni) {
            const bf16* s = &Bs[cur][(wn * 64 + ni * 16 + l16) * 32 + half * 8];
            *((int4*)&bfm[ni])     = *(const int4*)(s);
            *((int4*)&bfm[ni] + 1) = *(const int4*)(s + 16);
        }
#pragma unroll
        for (int mi = 0; mi < 2; ++mi)
#pragma unroll
            for (int ni = 0; ni < 4; ++ni)
                c[mi][ni] = __builtin_amdgcn_wmma_f32_16x16x32_bf16(
                    false, af[mi], false, bfm[ni], (short)0, c[mi][ni], false, false);

        // all waves' async writes must land before anyone reads the buffer
        asm volatile("s_wait_asynccnt 0x0" ::: "memory");
        __syncthreads();
    }

    // epilogue: C layout VGPR r -> M = r + 8*half, N = l16
#pragma unroll
    for (int mi = 0; mi < 2; ++mi)
#pragma unroll
        for (int ni = 0; ni < 4; ++ni)
#pragma unroll
            for (int r = 0; r < 8; ++r) {
                int row = bm + wm * 32 + mi * 16 + r + 8 * half;
                int col = bn + wn * 64 + ni * 16 + l16;
                float acc = c[mi][ni][r];
                size_t o = (size_t)row * N + col;
                if (mode == CT_MODE_F32) {
                    outF[o] = acc;
                } else if (mode == CT_MODE_RESID) {
                    outF[o] = aux[o] + scale[col] * acc;
                } else { // SWIGLU: acc is g3, aux is g1
                    float g = aux[o];
                    float sw = g / (1.f + __expf(-g));
                    outH[o] = f2bf(sw * acc);
                }
            }
}

// ------------- attention pass 1: windowed scores + softmax -> probs --------
__global__ __launch_bounds__(256)
void ct_attn_probs_kernel(const float* __restrict__ q, const float* __restrict__ k,
                          float* __restrict__ probs)
{
    int idx = blockIdx.x * 256 + threadIdx.x;  // over B*H*S = 65536
    int i = idx & (SEQ - 1);
    int h = (idx >> 11) & (NHEAD - 1);
    int b = idx >> 14;
    const float* qr = q + ((size_t)(b * SEQ + i)) * DIM + h * HDIM;
    const float slope = exp2f(-(float)(h + 1));   // 2^(-8/8)=0.5 -> 0.5^(h+1)
    const float inv_sqrt_d = 0.08838834764831845f;

    float sc[WIN + 1];
    float m = -1e30f;
#pragma unroll
    for (int t = 0; t <= WIN; ++t) {
        int j = i - WIN + t;
        float s = -1e30f;
        if (j >= 0) {
            const float* kr = k + ((size_t)(b * SEQ + j)) * DIM + h * HDIM;
            float acc = 0.f;
#pragma unroll 1
            for (int d = 0; d < HDIM; d += 4) {
                float4 a  = *(const float4*)(qr + d);
                float4 bb = *(const float4*)(kr + d);
                acc += a.x * bb.x + a.y * bb.y + a.z * bb.z + a.w * bb.w;
            }
            s = acc * inv_sqrt_d + slope * (float)(j - i);
        }
        sc[t] = s;
        m = fmaxf(m, s);
    }
    float l = 0.f;
#pragma unroll
    for (int t = 0; t <= WIN; ++t) {
        float e = (sc[t] > -1e29f) ? __expf(sc[t] - m) : 0.f;
        sc[t] = e;
        l += e;
    }
    float inv = 1.f / l;
    float* p = probs + (size_t)idx * (WIN + 1);
#pragma unroll
    for (int t = 0; t <= WIN; ++t) p[t] = sc[t] * inv;
}

// ------------- attention pass 2: y = P @ V, write bf16 ---------------------
__global__ __launch_bounds__(256)
void ct_attn_av_kernel(const float* __restrict__ v, const float* __restrict__ probs,
                       bf16* __restrict__ y)
{
    int idx = blockIdx.x * 256 + threadIdx.x;  // over B*H*S*32
    int g = idx & 31;
    int i = (idx >> 5) & (SEQ - 1);
    int h = (idx >> 16) & (NHEAD - 1);
    int b = idx >> 19;
    const float* p = probs + ((size_t)((b * NHEAD + h) * SEQ + i)) * (WIN + 1);
    float4 acc = make_float4(0.f, 0.f, 0.f, 0.f);
#pragma unroll 1
    for (int t = 0; t <= WIN; ++t) {
        int j = i - WIN + t;
        if (j >= 0) {
            float4 vv = *(const float4*)(v + ((size_t)(b * SEQ + j)) * DIM + h * HDIM + g * 4);
            float pt = p[t];
            acc.x += pt * vv.x; acc.y += pt * vv.y;
            acc.z += pt * vv.z; acc.w += pt * vv.w;
        }
    }
    size_t o = ((size_t)(b * SEQ + i)) * DIM + h * HDIM + g * 4;
    y[o + 0] = f2bf(acc.x); y[o + 1] = f2bf(acc.y);
    y[o + 2] = f2bf(acc.z); y[o + 3] = f2bf(acc.w);
}

// ---------------------------- launcher -------------------------------------
extern "C" void kernel_launch(void* const* d_in, const int* in_sizes, int n_in,
                              void* d_out, int out_size, void* d_ws, size_t ws_size,
                              hipStream_t stream)
{
    (void)in_sizes; (void)n_in; (void)out_size; (void)ws_size;
    const float* x       = (const float*)d_in[0];
    const float* wq      = (const float*)d_in[1];
    const float* wk      = (const float*)d_in[2];
    const float* wv      = (const float*)d_in[3];
    const float* wo      = (const float*)d_in[4];
    const float* qnw     = (const float*)d_in[5];
    const float* knw     = (const float*)d_in[6];
    const float* w1      = (const float*)d_in[7];
    const float* w2      = (const float*)d_in[8];
    const float* w3      = (const float*)d_in[9];
    const float* anw     = (const float*)d_in[10];
    const float* fnw     = (const float*)d_in[11];
    const float* asc     = (const float*)d_in[12];
    const float* fsc     = (const float*)d_in[13];
    float* out = (float*)d_out;   // residual stream lives here

    // workspace layout (bytes)
    char* ws = (char*)d_ws;
    const size_t MB = 1024ull * 1024ull;
    bf16*  xn     = (bf16*) (ws + 0);          //  16 MB : [8192,1024] bf16
    bf16*  wt     = (bf16*) (ws + 16 * MB);    //   8 MB : max [4096,1024] bf16
    float* qf     = (float*)(ws + 24 * MB);    //  32 MB
    float* kf     = (float*)(ws + 56 * MB);    //  32 MB
    float* vf     = (float*)(ws + 88 * MB);    //  32 MB
    float* probs  = (float*)(ws + 120 * MB);   // ~4.5 MB
    bf16*  ybf    = (bf16*) (ws + 126 * MB);   //  16 MB
    float* g1     = (float*)(ws + 142 * MB);   // 128 MB : [8192,4096] f32
    bf16*  hidden = (bf16*) (ws + 270 * MB);   //  64 MB : [8192,4096] bf16

    // residual = x
    hipMemcpyAsync(out, x, (size_t)ROWS * DIM * sizeof(float),
                   hipMemcpyDeviceToDevice, stream);

    const dim3 blk(256);
    const dim3 gemmD(DIM / 128, ROWS / 128);    // N=1024
    const dim3 gemmF(FFDIM / 128, ROWS / 128);  // N=4096
    const int cgD  = (DIM * DIM + 255) / 256;      // 1M elem weight convert
    const int cgFF = (DIM * FFDIM + 255) / 256;    // 4M elem weight convert

    for (int l = 0; l < 2; ++l) {
        const size_t wOffD  = (size_t)l * DIM * DIM;
        const size_t wOffFF = (size_t)l * DIM * FFDIM;
        const size_t vOff   = (size_t)l * DIM;

        // ---- attention ----
        ct_rmsnorm_kernel<<<ROWS, blk, 0, stream>>>(out, anw + vOff, 0.01f,
                                                    nullptr, xn, DIM);
        ct_convt_kernel<<<cgD, blk, 0, stream>>>(wq + wOffD, wt, DIM, DIM);
        ct_gemm_bf16_kernel<<<gemmD, blk, 0, stream>>>(xn, wt, ROWS, DIM, DIM,
            CT_MODE_F32, nullptr, nullptr, qf, nullptr);
        ct_convt_kernel<<<cgD, blk, 0, stream>>>(wk + wOffD, wt, DIM, DIM);
        ct_gemm_bf16_kernel<<<gemmD, blk, 0, stream>>>(xn, wt, ROWS, DIM, DIM,
            CT_MODE_F32, nullptr, nullptr, kf, nullptr);
        ct_convt_kernel<<<cgD, blk, 0, stream>>>(wv + wOffD, wt, DIM, DIM);
        ct_gemm_bf16_kernel<<<gemmD, blk, 0, stream>>>(xn, wt, ROWS, DIM, DIM,
            CT_MODE_F32, nullptr, nullptr, vf, nullptr);

        // qk-norm over flat 1024 dim (eps 1e-6), in place
        ct_rmsnorm_kernel<<<ROWS, blk, 0, stream>>>(qf, qnw + vOff, 1e-6f,
                                                    qf, nullptr, DIM);
        ct_rmsnorm_kernel<<<ROWS, blk, 0, stream>>>(kf, knw + vOff, 1e-6f,
                                                    kf, nullptr, DIM);

        ct_attn_probs_kernel<<<(BATCH * NHEAD * SEQ) / 256, blk, 0, stream>>>(qf, kf, probs);
        ct_attn_av_kernel<<<(BATCH * NHEAD * SEQ * 32) / 256, blk, 0, stream>>>(vf, probs, ybf);

        ct_convt_kernel<<<cgD, blk, 0, stream>>>(wo + wOffD, wt, DIM, DIM);
        ct_gemm_bf16_kernel<<<gemmD, blk, 0, stream>>>(ybf, wt, ROWS, DIM, DIM,
            CT_MODE_RESID, out, asc + vOff, out, nullptr);

        // ---- FFN (SwiGLU) ----
        ct_rmsnorm_kernel<<<ROWS, blk, 0, stream>>>(out, fnw + vOff, 0.01f,
                                                    nullptr, xn, DIM);
        ct_convt_kernel<<<cgFF, blk, 0, stream>>>(w1 + wOffFF, wt, DIM, FFDIM);
        ct_gemm_bf16_kernel<<<gemmF, blk, 0, stream>>>(xn, wt, ROWS, FFDIM, DIM,
            CT_MODE_F32, nullptr, nullptr, g1, nullptr);
        ct_convt_kernel<<<cgFF, blk, 0, stream>>>(w3 + wOffFF, wt, DIM, FFDIM);
        ct_gemm_bf16_kernel<<<gemmF, blk, 0, stream>>>(xn, wt, ROWS, FFDIM, DIM,
            CT_MODE_SWIGLU, g1, nullptr, nullptr, hidden);
        ct_convt_kernel<<<cgFF, blk, 0, stream>>>(w2 + wOffFF, wt, FFDIM, DIM);
        ct_gemm_bf16_kernel<<<gemmD, blk, 0, stream>>>(hidden, wt, ROWS, DIM, FFDIM,
            CT_MODE_RESID, out, fsc + vOff, out, nullptr);
    }
}
